// CausalSelfAttention_15023795602040
// MI455X (gfx1250) — matrix operand order repaired and checked by reading, not compiled, not run on hardware
//
#include <hip/hip_runtime.h>

typedef __attribute__((ext_vector_type(16))) __bf16 v16bf;
typedef __attribute__((ext_vector_type(8)))  float  v8f;
typedef unsigned short u16;
typedef int v4i_ __attribute__((vector_size(16)));
typedef __attribute__((address_space(1))) v4i_* gptr128;
typedef __attribute__((address_space(3))) v4i_* lptr128;

#define DEV __device__ __forceinline__

constexpr int   B_    = 2;
constexpr int   T_    = 2048;
constexpr int   DM    = 2048;          // model dim == hdim
constexpr int   H_    = 16;
constexpr int   DH    = 128;
constexpr int   NTOK  = B_ * T_;       // 4096
constexpr float EPS_  = 1.1920929e-07f;
constexpr float SCALE_ = 0.12f;

union FragU { v16bf v; uint4 u[2]; };

DEV u16 f2bf(float f) {
  unsigned u = __builtin_bit_cast(unsigned, f);
  u += 0x7FFFu + ((u >> 16) & 1u);     // round-to-nearest-even
  return (u16)(u >> 16);
}

// A-operand fragment: 16x32 bf16, rows striped lane&15, K split per ISA layout:
// lanes 0-15: K = k0+0..7 , k0+16..23 ; lanes 16-31: K = k0+8..15, k0+24..31
DEV v16bf load_frag_a(const u16* base, int ld, int k0) {
  const int lane = threadIdx.x & 31;
  const int r  = lane & 15;
  const int ko = k0 + ((lane & 16) >> 1);           // +8 for upper half
  FragU f;
  f.u[0] = *(const uint4*)(base + (size_t)r * ld + ko);
  f.u[1] = *(const uint4*)(base + (size_t)r * ld + ko + 16);
  return f.v;
}

// B-operand fragment: 32x16 bf16 (KxN), column n = lane&15, K split per ISA layout:
// lanes 0-15: K = k0+0..7 , k0+16..23 ; lanes 16-31: K = k0+8..15, k0+24..31
DEV v16bf load_frag_b(const u16* base, int ld, int k0) {
  const int lane = threadIdx.x & 31;
  const int r  = lane & 15;
  const int ko = k0 + ((lane & 16) >> 1);           // +8 for upper half
  FragU f;
  f.u[0] = *(const uint4*)(base + (size_t)r * ld + ko);
  f.u[1] = *(const uint4*)(base + (size_t)r * ld + ko + 16);
  return f.v;
}

DEV v8f wmma_bf16(v16bf a, v16bf b, v8f c) {
  return __builtin_amdgcn_wmma_f32_16x16x32_bf16(false, a, false, b,
                                                 (short)0, c, false, false);
}

// wave-local LDS store->load ordering fence (DS ops are in-order per wave).
DEV void lds_fence() {
  __builtin_amdgcn_wave_barrier();
  asm volatile("s_wait_dscnt 0" ::: "memory");
  __builtin_amdgcn_wave_barrier();
}

// -------- gfx1250 async Global->LDS copy (ASYNCcnt-tracked), with fallback --
DEV void async_copy_b128(const u16* src, u16* dst) {
#if __has_builtin(__builtin_amdgcn_global_load_async_to_lds_b128)
  __builtin_amdgcn_global_load_async_to_lds_b128(
      (gptr128)(void*)const_cast<u16*>(src),
      (lptr128)(void*)dst, 0, 0);
#else
  *(uint4*)dst = *(const uint4*)src;               // sync fallback
#endif
}

DEV void wait_async_le1() {
#if __has_builtin(__builtin_amdgcn_s_wait_asynccnt)
  __builtin_amdgcn_s_wait_asynccnt(1);
#elif __has_builtin(__builtin_amdgcn_global_load_async_to_lds_b128)
  asm volatile("s_wait_asynccnt 1" ::: "memory");
#endif
}
DEV void wait_async_le0() {
#if __has_builtin(__builtin_amdgcn_s_wait_asynccnt)
  __builtin_amdgcn_s_wait_asynccnt(0);
#elif __has_builtin(__builtin_amdgcn_global_load_async_to_lds_b128)
  asm volatile("s_wait_asynccnt 0" ::: "memory");
#endif
}

// ---------------------------------------------------------------- convert
__global__ void k_cvt_bf16(const float* __restrict__ src, u16* __restrict__ dst, int n4) {
  for (int i = blockIdx.x * blockDim.x + threadIdx.x; i < n4; i += gridDim.x * blockDim.x) {
    float4 f = ((const float4*)src)[i];
    ushort4 o;
    o.x = f2bf(f.x); o.y = f2bf(f.y); o.z = f2bf(f.z); o.w = f2bf(f.w);
    ((ushort4*)dst)[i] = o;
  }
}

// ------------------------------------------------- fused QKV GEMM + norm/rope
// grid: (NTOK/16, H, 3)  block: 256 (8 waves). Each block produces one
// 16-token x 128 (one head) slab of Q, K, or V. The activation A-slab is
// staged to LDS once per block via async Global->LDS DMA (double-buffered);
// each wave reads its B columns (weights) straight from global.
__global__ __launch_bounds__(256)
void k_qkv(const u16* __restrict__ xh,
           const u16* __restrict__ wq, const u16* __restrict__ wk, const u16* __restrict__ wv,
           const float* __restrict__ bq, const float* __restrict__ bk, const float* __restrict__ bv,
           u16* __restrict__ qh, u16* __restrict__ kh, u16* __restrict__ vt) {
  constexpr int ALD = 136;                       // padded bf16 row stride
  __shared__ __align__(16) u16 s_a[2][16 * ALD];
  __shared__ __align__(16) float s_tile[16][DH + 4];
  __shared__ float s_red[16][16];
  __shared__ float s_rstd[16];

  const int tok0 = blockIdx.x * 16;
  const int h    = blockIdx.y;
  const int z    = blockIdx.z;           // 0=Q 1=K 2=V
  const int wv32 = threadIdx.x >> 5;     // wave id 0..7
  const int lane = threadIdx.x & 31;
  const int half8 = (lane & 16) >> 1;

  const u16*   W    = (z == 0) ? wq : (z == 1) ? wk : wv;
  const float* bias = (z == 0) ? bq : (z == 1) ? bk : bv;

  const int n0 = h * DH + wv32 * 16;     // column in hdim
  const u16* Abase = xh + (size_t)tok0 * DM;
  const u16* Bbase = W  + (size_t)n0  * DM;

  // async stage: 16 rows x 128 K-cols = 256 x 16B chunks, one per thread
  const int crow = threadIdx.x >> 4;     // 0..15
  const int cchk = threadIdx.x & 15;     // 0..15 (8 bf16 each)

  v8f c = {};
  int buf = 0;
  async_copy_b128(Abase + (size_t)crow * DM + cchk * 8, &s_a[0][crow * ALD + cchk * 8]);
  for (int kk0 = 0; kk0 < DM; kk0 += 128) {
    const bool more = (kk0 + 128) < DM;
    if (more)
      async_copy_b128(Abase + (size_t)crow * DM + kk0 + 128 + cchk * 8,
                      &s_a[buf ^ 1][crow * ALD + cchk * 8]);
    if (more) wait_async_le1(); else wait_async_le0();
    __syncthreads();
    __builtin_prefetch(Bbase + (size_t)(lane & 15) * DM + kk0 + 256, 0, 3);
#pragma unroll
    for (int k0 = 0; k0 < 128; k0 += 32) {
      v16bf a = load_frag_a(&s_a[buf][0], ALD, k0);
      v16bf b = load_frag_b(Bbase, DM, kk0 + k0);
      c = wmma_bf16(a, b, c);
    }
    __syncthreads();                     // everyone done reading buf
    buf ^= 1;
  }

  const float bval = bias[n0 + (lane & 15)];
#pragma unroll
  for (int r = 0; r < 8; ++r)
    s_tile[r + half8][wv32 * 16 + (lane & 15)] = c[r] + bval;
  __syncthreads();

  // per-row sum of squares over the 128-wide head (RMS norm)
  {
    const int row = threadIdx.x >> 4, seg = threadIdx.x & 15;
    float p = 0.f;
#pragma unroll
    for (int cc = 0; cc < 8; ++cc) {
      float v = s_tile[row][seg * 8 + cc];
      p += v * v;
    }
    s_red[row][seg] = p;
  }
  __syncthreads();
  if (threadIdx.x < 16) {
    float s = 0.f;
#pragma unroll
    for (int i = 0; i < 16; ++i) s += s_red[threadIdx.x][i];
    s_rstd[threadIdx.x] = rsqrtf(s * (1.0f / DH) + EPS_);
  }
  __syncthreads();

  const int b_idx = tok0 / T_;
  const int trow0 = tok0 % T_;
  const int bh    = b_idx * H_ + h;

  if (z < 2) {                           // Q / K : rmsnorm + rotary, row-major
    u16* dst = (z == 0) ? qh : kh;
#pragma unroll
    for (int it = 0; it < 4; ++it) {
      int idx = threadIdx.x + it * 256;  // 16 rows x 64 pairs
      int row = idx >> 6, j = idx & 63;
      float rs = s_rstd[row];
      float x1 = s_tile[row][j] * rs;
      float x2 = s_tile[row][j + 64] * rs;
      float fr = (j < 32) ? exp2f(-10.0f * (float)j * (1.0f / 31.0f)) : 0.0f;
      float th = (float)(trow0 + row) * fr;
      float cs = cosf(th), sn = sinf(th);
      float y1 = x1 * cs + x2 * sn;
      float y2 = x2 * cs - x1 * sn;
      size_t base = ((size_t)bh * T_ + (trow0 + row)) * DH;
      dst[base + j]      = f2bf(y1);
      dst[base + j + 64] = f2bf(y2);
    }
  } else {                               // V : store transposed [bh][d][T]
#pragma unroll
    for (int it = 0; it < 8; ++it) {
      int idx = threadIdx.x + it * 256;  // 16 rows x 128 cols
      int row = idx >> 7, d = idx & 127;
      vt[((size_t)bh * DH + d) * T_ + (trow0 + row)] = f2bf(s_tile[row][d]);
    }
  }
}

// ------------------------------------------------------- flash attention
// grid: B*H*(T/128) blocks of 256; each wave owns a 16-row q tile.
__global__ __launch_bounds__(256)
void k_attn(const u16* __restrict__ qh, const u16* __restrict__ kh,
            const u16* __restrict__ vt, u16* __restrict__ yh) {
  constexpr int PLD = 40;                // padded bf16 row stride for P scratch
  __shared__ __align__(16) u16 p_lds[8][16 * PLD];

  const int qpb  = T_ / 128;             // q blocks per (b,h)
  const int bh   = blockIdx.x / qpb;
  const int qblk = blockIdx.x % qpb;
  const int wv32 = threadIdx.x >> 5;
  const int lane = threadIdx.x & 31;
  const int half8 = (lane & 16) >> 1;

  const int q0 = qblk * 128 + wv32 * 16;
  const int b_idx = bh / H_, h = bh % H_;

  const u16* Qbh = qh + (size_t)bh * T_ * DH;
  const u16* Kbh = kh + (size_t)bh * T_ * DH;
  const u16* Vbh = vt + (size_t)bh * DH * T_;

  v16bf qf[4];
#pragma unroll
  for (int dd = 0; dd < 4; ++dd)
    qf[dd] = load_frag_a(Qbh + (size_t)q0 * DH, DH, dd * 32);

  v8f o[8];
#pragma unroll
  for (int ds = 0; ds < 8; ++ds) o[ds] = (v8f){};
  float mrow[8], lrow[8];
#pragma unroll
  for (int r = 0; r < 8; ++r) { mrow[r] = -1e30f; lrow[r] = 0.f; }

  u16* pl = &p_lds[wv32][0];
  const int kv_end = q0 + 16;            // per-wave causal bound

  for (int kv0 = 0; kv0 < kv_end; kv0 += 32) {
    // prefetch next kv tile's K rows (WGP scope)
    __builtin_prefetch(Kbh + (size_t)(kv0 + 32 + (lane & 15)) * DH, 0, 3);
    v8f s0 = {}, s1 = {};
#pragma unroll
    for (int dd = 0; dd < 4; ++dd) {
      v16bf k0f = load_frag_b(Kbh + (size_t)kv0 * DH,        DH, dd * 32);
      v16bf k1f = load_frag_b(Kbh + (size_t)(kv0 + 16) * DH, DH, dd * 32);
      s0 = wmma_bf16(qf[dd], k0f, s0);
      s1 = wmma_bf16(qf[dd], k1f, s1);
    }
    // scale + causal mask (C layout: row = r+half8, col = lane&15 within tile)
#pragma unroll
    for (int r = 0; r < 8; ++r) {
      int m_g  = q0 + half8 + r;
      int n0_g = kv0 + (lane & 15);
      s0[r] = (n0_g      <= m_g) ? s0[r] * SCALE_ : -1e30f;
      s1[r] = (n0_g + 16 <= m_g) ? s1[r] * SCALE_ : -1e30f;
    }
    float alpha[8];
#pragma unroll
    for (int r = 0; r < 8; ++r) {
      float v = fmaxf(s0[r], s1[r]);
      v = fmaxf(v, __shfl_xor(v, 8));
      v = fmaxf(v, __shfl_xor(v, 4));
      v = fmaxf(v, __shfl_xor(v, 2));
      v = fmaxf(v, __shfl_xor(v, 1));
      float mn = fmaxf(mrow[r], v);
      alpha[r] = __expf(mrow[r] - mn);
      mrow[r] = mn;
      s0[r] = __expf(s0[r] - mn);
      s1[r] = __expf(s1[r] - mn);
      float t = s0[r] + s1[r];
      t += __shfl_xor(t, 8);
      t += __shfl_xor(t, 4);
      t += __shfl_xor(t, 2);
      t += __shfl_xor(t, 1);
      lrow[r] = lrow[r] * alpha[r] + t;
    }
    // repack P (C layout fp32) -> bf16 A layout via wave-private LDS scratch
#pragma unroll
    for (int r = 0; r < 8; ++r) {
      int row = r + half8;
      pl[row * PLD + (lane & 15)]      = f2bf(s0[r]);
      pl[row * PLD + 16 + (lane & 15)] = f2bf(s1[r]);
    }
    lds_fence();
    v16bf pf = load_frag_a(pl, PLD, 0);
#pragma unroll
    for (int ds = 0; ds < 8; ++ds) {
#pragma unroll
      for (int r = 0; r < 8; ++r) o[ds][r] *= alpha[r];
      v16bf vf = load_frag_b(Vbh + (size_t)(ds * 16) * T_, T_, kv0);
      o[ds] = wmma_bf16(pf, vf, o[ds]);
    }
  }

  float inv_l[8];
#pragma unroll
  for (int r = 0; r < 8; ++r) inv_l[r] = 1.0f / lrow[r];

  const size_t ybase = ((size_t)(b_idx * T_ + q0)) * DM + (size_t)h * DH;
#pragma unroll
  for (int ds = 0; ds < 8; ++ds)
#pragma unroll
    for (int r = 0; r < 8; ++r) {
      int row = r + half8;
      yh[ybase + (size_t)row * DM + ds * 16 + (lane & 15)] = f2bf(o[ds][r] * inv_l[r]);
    }
}

// ---------------------------------------------------------- output projection
// grid: (NTOK/16, DM/128) blocks of 256; out = Y @ Wp^T + bp (fp32 out).
// Same async-staged A-slab structure as k_qkv.
__global__ __launch_bounds__(256)
void k_oproj(const u16* __restrict__ yh, const u16* __restrict__ wp,
             const float* __restrict__ bp, float* __restrict__ out) {
  constexpr int ALD = 136;
  __shared__ __align__(16) u16 s_a[2][16 * ALD];

  const int tok0 = blockIdx.x * 16;
  const int wv32 = threadIdx.x >> 5;
  const int lane = threadIdx.x & 31;
  const int half8 = (lane & 16) >> 1;
  const int n0 = blockIdx.y * 128 + wv32 * 16;

  const u16* Abase = yh + (size_t)tok0 * DM;
  const u16* Bbase = wp + (size_t)n0  * DM;

  const int crow = threadIdx.x >> 4;
  const int cchk = threadIdx.x & 15;

  v8f c = {};
  int buf = 0;
  async_copy_b128(Abase + (size_t)crow * DM + cchk * 8, &s_a[0][crow * ALD + cchk * 8]);
  for (int kk0 = 0; kk0 < DM; kk0 += 128) {
    const bool more = (kk0 + 128) < DM;
    if (more)
      async_copy_b128(Abase + (size_t)crow * DM + kk0 + 128 + cchk * 8,
                      &s_a[buf ^ 1][crow * ALD + cchk * 8]);
    if (more) wait_async_le1(); else wait_async_le0();
    __syncthreads();
    __builtin_prefetch(Bbase + (size_t)(lane & 15) * DM + kk0 + 256, 0, 3);
#pragma unroll
    for (int k0 = 0; k0 < 128; k0 += 32) {
      v16bf a = load_frag_a(&s_a[buf][0], ALD, k0);
      v16bf b = load_frag_b(Bbase, DM, kk0 + k0);
      c = wmma_bf16(a, b, c);
    }
    __syncthreads();
    buf ^= 1;
  }
  const float bval = bp[n0 + (lane & 15)];
#pragma unroll
  for (int r = 0; r < 8; ++r)
    out[((size_t)tok0 + r + half8) * DM + n0 + (lane & 15)] = c[r] + bval;
}

// -------------------------------------------------------------------- launch
extern "C" void kernel_launch(void* const* d_in, const int* in_sizes, int n_in,
                              void* d_out, int out_size, void* d_ws, size_t ws_size,
                              hipStream_t stream) {
  (void)in_sizes; (void)n_in; (void)out_size; (void)ws_size;
  const float* x  = (const float*)d_in[0];
  const float* Wq = (const float*)d_in[1];
  const float* bq = (const float*)d_in[2];
  const float* Wk = (const float*)d_in[3];
  const float* bk = (const float*)d_in[4];
  const float* Wv = (const float*)d_in[5];
  const float* bv = (const float*)d_in[6];
  const float* Wp = (const float*)d_in[7];
  const float* bp = (const float*)d_in[8];
  float* out = (float*)d_out;

  u16* ws = (u16*)d_ws;
  size_t off = 0;
  u16* xh  = ws + off; off += (size_t)NTOK * DM;   // 16 MB
  u16* wqh = ws + off; off += (size_t)DM * DM;     // 8 MB
  u16* wkh = ws + off; off += (size_t)DM * DM;
  u16* wvh = ws + off; off += (size_t)DM * DM;
  u16* wph = ws + off; off += (size_t)DM * DM;
  u16* qh  = ws + off; off += (size_t)NTOK * DM;   // [bh][t][128]
  u16* kh  = ws + off; off += (size_t)NTOK * DM;
  u16* vt  = ws + off; off += (size_t)NTOK * DM;   // [bh][128][T]
  u16* yh  = ws + off; off += (size_t)NTOK * DM;

  k_cvt_bf16<<<1024, 256, 0, stream>>>(x,  xh,  NTOK * DM / 4);
  k_cvt_bf16<<<512,  256, 0, stream>>>(Wq, wqh, DM * DM / 4);
  k_cvt_bf16<<<512,  256, 0, stream>>>(Wk, wkh, DM * DM / 4);
  k_cvt_bf16<<<512,  256, 0, stream>>>(Wv, wvh, DM * DM / 4);
  k_cvt_bf16<<<512,  256, 0, stream>>>(Wp, wph, DM * DM / 4);

  k_qkv<<<dim3(NTOK / 16, H_, 3), 256, 0, stream>>>(xh, wqh, wkh, wvh,
                                                    bq, bk, bv, qh, kh, vt);
  k_attn<<<dim3(B_ * H_ * (T_ / 128)), 256, 0, stream>>>(qh, kh, vt, yh);
  k_oproj<<<dim3(NTOK / 16, DM / 128), 256, 0, stream>>>(yh, wph, bp, out);
}
